// GraphSAGENet_3959959846912
// MI455X (gfx1250) — matrix operand order repaired and checked by reading, not compiled, not run on hardware
//
#include <hip/hip_runtime.h>
#include <hip/hip_bf16.h>

typedef __attribute__((ext_vector_type(16))) __bf16 v16bf;
typedef __attribute__((ext_vector_type(8)))  float  v8f;

#define NNODES 50000
#define NEDGES 400000
#define DIM    256

// ---------------------------------------------------------------------------
// Degree count: one thread per edge, integer atomic on dst.
// ---------------------------------------------------------------------------
__global__ __launch_bounds__(256) void sage_deg_kernel(
    const long long* __restrict__ ei, int* __restrict__ deg, int E)
{
    int e = blockIdx.x * blockDim.x + threadIdx.x;
    if (e < E) {
        int dst = (int)ei[E + e];
        atomicAdd(&deg[dst], 1);
    }
}

__global__ __launch_bounds__(256) void sage_invdeg_kernel(
    const int* __restrict__ deg, float* __restrict__ invd, int n)
{
    int i = blockIdx.x * blockDim.x + threadIdx.x;
    if (i < n) {
        invd[i] = 1.0f / fmaxf((float)deg[i], 1.0f);
    }
}

// ---------------------------------------------------------------------------
// One-time weight prep: transpose 256x256 fp32 W into column-major bf16
// Wt[n][k] = W[k][n]. 128KB per matrix -> permanently L2-hot. This turns the
// GEMM B-fragment (16 consecutive-K values of one column) into a single
// contiguous 32-byte read instead of a 1KB-stride 16-element gather.
// ---------------------------------------------------------------------------
__global__ __launch_bounds__(256) void prep_w_kernel(
    const float* __restrict__ W, __bf16* __restrict__ Wt)
{
    int t = blockIdx.x * blockDim.x + threadIdx.x;   // 0..65535
    int n = t >> 8;
    int k = t & 255;
    Wt[(size_t)n * DIM + k] = (__bf16)W[(size_t)k * DIM + n];
}

// ---------------------------------------------------------------------------
// Edge scatter: one wave32 per edge. Lane l moves 8 contiguous floats:
// two float4 loads of x[src] (1KB/wave, fully coalesced) and 8 HW float
// atomics into agg[dst]. x/agg are 51.2MB each -> L2 resident (192MB L2).
// Prefetch the next edge's source row (global_prefetch_b8).
// ---------------------------------------------------------------------------
__global__ __launch_bounds__(256) void sage_scatter_kernel(
    const float* __restrict__ x, const long long* __restrict__ ei,
    float* __restrict__ agg, int E)
{
    int lane   = threadIdx.x & 31;
    int wave   = (blockIdx.x * blockDim.x + threadIdx.x) >> 5;
    int nwaves = (gridDim.x * blockDim.x) >> 5;

    for (int e = wave; e < E; e += nwaves) {
        int src = (int)ei[e];
        int dst = (int)ei[E + e];

        int en = e + nwaves;
        if (en < E) {
            int nsrc = (int)ei[en];
            __builtin_prefetch(x + (size_t)nsrc * DIM + lane * 8, 0, 3);
        }

        const float4* xs = (const float4*)(x + (size_t)src * DIM);
        float4 v0 = xs[lane * 2 + 0];
        float4 v1 = xs[lane * 2 + 1];

        float* ag = agg + (size_t)dst * DIM + lane * 8;
        unsafeAtomicAdd(ag + 0, v0.x);
        unsafeAtomicAdd(ag + 1, v0.y);
        unsafeAtomicAdd(ag + 2, v0.z);
        unsafeAtomicAdd(ag + 3, v0.w);
        unsafeAtomicAdd(ag + 4, v1.x);
        unsafeAtomicAdd(ag + 5, v1.y);
        unsafeAtomicAdd(ag + 6, v1.z);
        unsafeAtomicAdd(ag + 7, v1.w);
    }
}

// ---------------------------------------------------------------------------
// Fused SAGE linear:  out = relu?( (agg*invd) @ Wl + xin @ Wr + bias )
// One wave32 per 16x16 output tile; fully unrolled K-loop -> 16 static
// v_wmma_f32_16x16x32_bf16 per tile with pipelined loads.
//
// A frag (16x32 bf16): lane holds row M = lane%16; klo = 16*(lane/16);
//   elems 0..15 -> K = klo..klo+15 (same contiguous order as B).
// B frag (32x16 bf16): lane holds col N = lane%16; elem j -> K = krow + j,
//   krow = 16*(lane/16); contiguous in the pre-transposed bf16 Wt.
// C frag (16x16 f32): vgpr r, lane -> M = r + 8*(lane/16), N = lane%16.
// ---------------------------------------------------------------------------
template <int RELU>
__global__ __launch_bounds__(256) void sage_gemm_kernel(
    const float* __restrict__ agg, const float* __restrict__ invd,
    const float* __restrict__ xin,
    const __bf16* __restrict__ Wtl, const __bf16* __restrict__ Wtr,
    const float* __restrict__ bias, float* __restrict__ out)
{
    const int wave = threadIdx.x >> 5;
    const int lane = threadIdx.x & 31;
    const int tile = blockIdx.x * 8 + wave;        // 0..49999
    if (tile >= (NNODES / 16) * (DIM / 16)) return;

    const int row_tile = tile >> 4;
    const int col_tile = tile & 15;
    const int rowbase  = row_tile * 16;
    const int colbase  = col_tile * 16;

    const int laneLo = lane & 15;
    const int laneHi = lane >> 4;
    const int m   = rowbase + laneLo;              // A row this lane feeds
    const int col = colbase + laneLo;              // B/C column this lane feeds
    const float inv = invd[m];

    const float*  ra  = agg + (size_t)m * DIM;
    const float*  rx  = xin + (size_t)m * DIM;
    const __bf16* pbl = Wtl + (size_t)col * DIM;   // column-major: contiguous K
    const __bf16* pbr = Wtr + (size_t)col * DIM;

    v8f c = {0.f, 0.f, 0.f, 0.f, 0.f, 0.f, 0.f, 0.f};

#pragma unroll
    for (int kb = 0; kb < DIM; kb += 32) {
        const int klo  = kb + (laneHi << 4);
        const int krow = kb + (laneHi << 4);

        v16bf a1, a2;
#pragma unroll
        for (int j = 0; j < 8; ++j) {
            a1[j]     = (__bf16)(ra[klo + j] * inv);
            a1[j + 8] = (__bf16)(ra[klo + 8 + j] * inv);
            a2[j]     = (__bf16)(rx[klo + j]);
            a2[j + 8] = (__bf16)(rx[klo + 8 + j]);
        }
        // 32-byte aligned contiguous B fragments (2x global_load_b128 each)
        v16bf bl = *(const v16bf*)(pbl + krow);
        v16bf br = *(const v16bf*)(pbr + krow);

        c = __builtin_amdgcn_wmma_f32_16x16x32_bf16(
                false, a1, false, bl, (short)0, c, false, false);
        c = __builtin_amdgcn_wmma_f32_16x16x32_bf16(
                false, a2, false, br, (short)0, c, false, false);
    }

    const float bv = bias[col];
#pragma unroll
    for (int r = 0; r < 8; ++r) {
        float v = c[r] + bv;
        if (RELU) v = fmaxf(v, 0.0f);
        out[(size_t)(rowbase + r + (laneHi << 3)) * DIM + col] = v;
    }
}

// ---------------------------------------------------------------------------
// Launch: prep W -> deg -> invdeg -> [zero agg, scatter, gemm] x 2
// ---------------------------------------------------------------------------
extern "C" void kernel_launch(void* const* d_in, const int* in_sizes, int n_in,
                              void* d_out, int out_size, void* d_ws, size_t ws_size,
                              hipStream_t stream) {
    const float*     x   = (const float*)d_in[0];
    const long long* ei  = (const long long*)d_in[1];   // int64 per reference
    const float*     W1l = (const float*)d_in[2];
    const float*     W1r = (const float*)d_in[3];
    const float*     b1  = (const float*)d_in[4];
    const float*     W2l = (const float*)d_in[5];
    const float*     W2r = (const float*)d_in[6];
    const float*     b2  = (const float*)d_in[7];
    float*           out = (float*)d_out;

    // Workspace layout (512B aligned blocks)
    char*   w     = (char*)d_ws;
    size_t  off   = 0;
    int*    deg   = (int*)(w + off);    off += ((size_t)NNODES * 4 + 511) & ~511ULL;
    float*  invd  = (float*)(w + off);  off += ((size_t)NNODES * 4 + 511) & ~511ULL;
    float*  agg   = (float*)(w + off);  off += ((size_t)NNODES * DIM * 4 + 511) & ~511ULL;
    float*  h     = (float*)(w + off);  off += ((size_t)NNODES * DIM * 4 + 511) & ~511ULL;
    __bf16* Wt1l  = (__bf16*)(w + off); off += (size_t)DIM * DIM * 2;
    __bf16* Wt1r  = (__bf16*)(w + off); off += (size_t)DIM * DIM * 2;
    __bf16* Wt2l  = (__bf16*)(w + off); off += (size_t)DIM * DIM * 2;
    __bf16* Wt2r  = (__bf16*)(w + off); off += (size_t)DIM * DIM * 2;
    (void)ws_size; (void)in_sizes; (void)n_in; (void)out_size;

    const size_t aggBytes = (size_t)NNODES * DIM * sizeof(float);

    // Weight prep (bf16 transpose, L2-hot thereafter)
    prep_w_kernel<<<(DIM * DIM) / 256, 256, 0, stream>>>(W1l, Wt1l);
    prep_w_kernel<<<(DIM * DIM) / 256, 256, 0, stream>>>(W1r, Wt1r);
    prep_w_kernel<<<(DIM * DIM) / 256, 256, 0, stream>>>(W2l, Wt2l);
    prep_w_kernel<<<(DIM * DIM) / 256, 256, 0, stream>>>(W2r, Wt2r);

    // Degree (shared by both layers)
    hipMemsetAsync(deg, 0, (size_t)NNODES * sizeof(int), stream);
    sage_deg_kernel<<<(NEDGES + 255) / 256, 256, 0, stream>>>(ei, deg, NEDGES);
    sage_invdeg_kernel<<<(NNODES + 255) / 256, 256, 0, stream>>>(deg, invd, NNODES);

    // Layer 1
    hipMemsetAsync(agg, 0, aggBytes, stream);
    sage_scatter_kernel<<<6250, 256, 0, stream>>>(x, ei, agg, NEDGES);
    sage_gemm_kernel<1><<<6250, 256, 0, stream>>>(agg, invd, x, Wt1l, Wt1r, b1, h);

    // Layer 2
    hipMemsetAsync(agg, 0, aggBytes, stream);
    sage_scatter_kernel<<<6250, 256, 0, stream>>>(h, ei, agg, NEDGES);
    sage_gemm_kernel<0><<<6250, 256, 0, stream>>>(agg, invd, h, Wt2l, Wt2r, b2, out);
}
